// RCC_Attention_19902878450001
// MI455X (gfx1250) — matrix operand order repaired and checked by reading, not compiled, not run on hardware
//
#include <hip/hip_runtime.h>

typedef __bf16 bf16;
typedef __attribute__((ext_vector_type(16))) __bf16 v16bf;
typedef __attribute__((ext_vector_type(8)))  __bf16 v8bf;
typedef __attribute__((ext_vector_type(4)))  __bf16 v4bf;
typedef __attribute__((ext_vector_type(8)))  float  v8f;

constexpr int B = 8, C = 192, CQK = 64, H = 128, W = 128;
constexpr int HW = H * W;
constexpr int NATT = 256;          // H + W logits per query pixel

// ---- wave32 WMMA fragment loaders (CDNA5 ISA 7.12.2, 16-bit operands) ----
// All LDS layouts below keep the k-dimension contiguous per fragment owner,
// so each fragment is two aligned 16-byte ds_load_b128s.
union Frag { v16bf v; v8bf h[2]; };

// A 16x32: lane L owns row (L&15); k = e + (e<8?0:8) + (L<16?0:8)
__device__ __forceinline__ v16bf frag_a(const bf16* row, int lane) {
  const int kb = (lane < 16) ? 0 : 8;
  Frag f;
  f.h[0] = *(const v8bf*)(row + kb);
  f.h[1] = *(const v8bf*)(row + 16 + kb);
  return f.v;
}
// B 32x16: lane L owns col (L&15); k = e + (e<8?0:8) + (L<16?0:8)
__device__ __forceinline__ v16bf frag_b(const bf16* col, int lane) {
  const int ko = (lane < 16) ? 0 : 8;
  Frag f;
  f.h[0] = *(const v8bf*)(col + ko);
  f.h[1] = *(const v8bf*)(col + ko + 16);
  return f.v;
}

__device__ __forceinline__ v8f wmma_bf16(v16bf a, v16bf b, v8f c) {
  return __builtin_amdgcn_wmma_f32_16x16x32_bf16(false, a, false, b,
                                                 (short)0, c, false, false);
}

// ------------------------------------------------------------------------
// Kernel 1: fused Q/K/V 1x1-conv projection.
//   out[oc, p] = sum_c W[oc, c] * x[b, c, p] + bias[oc],  oc in [0,320)
//   Q/K/V stored position-major bf16: Qp/Kp [B][HW][64], Vp [B][HW][192]
// Block: 256 thr (8 waves as 2x4 tile grid) -> 32 out-ch x 64 positions.
// ------------------------------------------------------------------------
__global__ __launch_bounds__(256)
void qkv_proj_kernel(const float* __restrict__ x,
                     const float* __restrict__ Wq, const float* __restrict__ bq,
                     const float* __restrict__ Wk, const float* __restrict__ bk,
                     const float* __restrict__ Wv, const float* __restrict__ bv,
                     bf16* __restrict__ Qp, bf16* __restrict__ Kp,
                     bf16* __restrict__ Vp) {
  __shared__ bf16 ldsW[32][C + 8];     // A operand: [oc][c], c contiguous
  __shared__ bf16 ldsXt[64][C + 8];    // B^T operand: [p][c], c contiguous

  const int tid = threadIdx.x;
  const int b   = blockIdx.z;
  const int oc0 = blockIdx.y * 32;     // 0..288
  const int hw0 = blockIdx.x * 64;

  // stage x tile transposed: wide coalesced global reads, scattered b16 stores
  for (int i = tid; i < C * 16; i += 256) {      // 16 float4 groups per channel
    int c = i >> 4, p = (i & 15) * 4;
    const float4 xv = *(const float4*)&x[((size_t)b * C + c) * HW + hw0 + p];
    ldsXt[p + 0][c] = (bf16)xv.x;
    ldsXt[p + 1][c] = (bf16)xv.y;
    ldsXt[p + 2][c] = (bf16)xv.z;
    ldsXt[p + 3][c] = (bf16)xv.w;
  }
  for (int i = tid; i < 32 * C; i += 256) {
    int r = i / C, c = i % C;
    int gr = oc0 + r;
    const float* wrow = (gr < CQK)     ? (Wq + (size_t)gr * C)
                      : (gr < 2 * CQK) ? (Wk + (size_t)(gr - CQK) * C)
                                       : (Wv + (size_t)(gr - 2 * CQK) * C);
    ldsW[r][c] = (bf16)wrow[c];
  }
  __syncthreads();

  const int wave = tid >> 5, lane = tid & 31;
  const int m0 = (wave >> 2) * 16;     // out-channel sub-tile
  const int p0 = (wave & 3) * 16;      // position sub-tile
  const int mrow = lane & 15;
  const bool hi = lane >= 16;

  v8f acc;
  #pragma unroll
  for (int r = 0; r < 8; ++r) {
    int ocg = oc0 + m0 + r + (hi ? 8 : 0);
    acc[r] = (ocg < CQK)     ? bq[ocg]
           : (ocg < 2 * CQK) ? bk[ocg - CQK]
                             : bv[ocg - 2 * CQK];
  }

  #pragma unroll
  for (int kk = 0; kk < C; kk += 32) {
    v16bf a  = frag_a(&ldsW[m0 + mrow][kk], lane);
    v16bf bb = frag_b(&ldsXt[p0 + mrow][kk], lane);
    acc = wmma_bf16(a, bb, acc);
  }

  v8bf pk;
  #pragma unroll
  for (int r = 0; r < 8; ++r) pk[r] = (bf16)acc[r];

  const int ocg0 = oc0 + m0 + (hi ? 8 : 0);      // multiple of 8, one region
  const size_t pos = (size_t)b * HW + hw0 + p0 + mrow;
  bf16* dst; int cloc, cs;
  if (ocg0 < CQK)          { dst = Qp; cloc = ocg0;           cs = CQK; }
  else if (ocg0 < 2 * CQK) { dst = Kp; cloc = ocg0 - CQK;     cs = CQK; }
  else                     { dst = Vp; cloc = ocg0 - 2 * CQK; cs = C;   }
  *(v8bf*)(dst + pos * cs + cloc) = pk;          // 16B aligned store
}

// ------------------------------------------------------------------------
// Kernel 2: Gram matrices. MODE 0: per (b,w) column, E = Q_colT K_col with
// -inf diagonal, written to Att[..][0:128]. MODE 1: per (b,h) row, written
// to Att[..][128:256]. One block per (b,fix); each wave owns a 16x128 strip.
// Both operands live as [row/col][c] in LDS: k-contiguous for A and B.
// ------------------------------------------------------------------------
template <int MODE>
__global__ __launch_bounds__(256)
void gram_kernel(const bf16* __restrict__ Qp, const bf16* __restrict__ Kp,
                 bf16* __restrict__ Att) {
  __shared__ bf16 lq[H][CQK + 8];
  __shared__ bf16 lk[H][CQK + 8];

  const int tid = threadIdx.x;
  const int b = blockIdx.y, fix = blockIdx.x;

  const size_t pbase = (size_t)b * HW + (MODE == 0 ? (size_t)fix : (size_t)fix * W);
  const size_t qbase = pbase * CQK;
  const size_t qstr  = (MODE == 0) ? (size_t)W * CQK : (size_t)CQK;
  const size_t abase = pbase * NATT + (MODE == 0 ? 0 : H);
  const size_t astr  = (MODE == 0) ? (size_t)W * NATT : (size_t)NATT;

  // 8 contiguous bf16 per thread item -> b128 global loads, b128 ds stores
  for (int i = tid; i < H * (CQK / 8); i += 256) {
    int r = i >> 3, c = (i & 7) * 8;
    *(v8bf*)&lq[r][c] = *(const v8bf*)&Qp[qbase + (size_t)r * qstr + c];
    *(v8bf*)&lk[r][c] = *(const v8bf*)&Kp[qbase + (size_t)r * qstr + c];
  }
  __syncthreads();

  const int wave = tid >> 5, lane = tid & 31;
  const int m0 = wave * 16;
  const int mrow = lane & 15;
  const bool hi = lane >= 16;

  for (int n0 = 0; n0 < H; n0 += 16) {
    v8f acc = {};
    #pragma unroll
    for (int kk = 0; kk < CQK; kk += 32) {
      v16bf a  = frag_a(&lq[m0 + mrow][kk], lane);
      v16bf bb = frag_b(&lk[n0 + mrow][kk], lane);
      acc = wmma_bf16(a, bb, acc);
    }
    #pragma unroll
    for (int r = 0; r < 8; ++r) {
      int row = m0 + r + (hi ? 8 : 0);
      int col = n0 + mrow;
      float v = acc[r];
      if (MODE == 0 && row == col) v = -__builtin_inff();
      Att[abase + (size_t)row * astr + col] = (bf16)v;
    }
  }
}

// ------------------------------------------------------------------------
// Kernel 3: softmax over each pixel's 256 logits, in place (bf16).
// One wave per row; lane holds 8 contiguous values (one 16B load/store).
// ------------------------------------------------------------------------
__global__ __launch_bounds__(256)
void softmax_kernel(bf16* __restrict__ Att) {
  const int wave = threadIdx.x >> 5, lane = threadIdx.x & 31;
  const size_t row = (size_t)blockIdx.x * 8 + wave;
  bf16* p = Att + row * NATT + lane * 8;

  v8bf v = *(const v8bf*)p;
  float f[8];
  float mx = -__builtin_inff();
  #pragma unroll
  for (int i = 0; i < 8; ++i) { f[i] = (float)v[i]; mx = fmaxf(mx, f[i]); }
  #pragma unroll
  for (int s = 16; s > 0; s >>= 1) mx = fmaxf(mx, __shfl_xor(mx, s, 32));
  float sum = 0.f;
  #pragma unroll
  for (int i = 0; i < 8; ++i) { f[i] = __expf(f[i] - mx); sum += f[i]; }
  #pragma unroll
  for (int s = 16; s > 0; s >>= 1) sum += __shfl_xor(sum, s, 32);
  const float inv = 1.0f / sum;
  #pragma unroll
  for (int i = 0; i < 8; ++i) v[i] = (bf16)(f[i] * inv);
  *(v8bf*)p = v;
}

// ------------------------------------------------------------------------
// Kernel 4: apply attention.  outT[row, c] = sum_k A[row, k] * VT[k, c].
// MODE 0 (per (b,w) column): writes Acc. MODE 1 (per (b,h) row): adds.
// A (128x128) staged fully in LDS row-major (k contiguous); V staged in
// 32-row K-slices TRANSPOSED (lVt[c][k], k contiguous) so B fragments are
// two b128 ds loads. Each wave: 16-row strip x 192 channels (12 acc tiles).
// ------------------------------------------------------------------------
template <int MODE>
__global__ __launch_bounds__(256)
void attn_out_kernel(const bf16* __restrict__ Att, const bf16* __restrict__ Vp,
                     float* __restrict__ Acc) {
  __shared__ bf16 lA[H][H + 8];        // 34.8 KB, [row][k]
  __shared__ bf16 lVt[C][32 + 8];      // 15.0 KB, [c][k-slice]

  const int tid = threadIdx.x;
  const int b = blockIdx.y, fix = blockIdx.x;

  const size_t pbase = (size_t)b * HW + (MODE == 0 ? (size_t)fix : (size_t)fix * W);
  const size_t abase = pbase * NATT + (MODE == 0 ? 0 : H);
  const size_t astr  = (MODE == 0) ? (size_t)W * NATT : (size_t)NATT;
  const size_t vbase = pbase * C;
  const size_t vstr  = (MODE == 0) ? (size_t)W * C : (size_t)C;

  for (int i = tid; i < H * (H / 8); i += 256) {
    int r = i >> 4, c = (i & 15) * 8;
    *(v8bf*)&lA[r][c] = *(const v8bf*)&Att[abase + (size_t)r * astr + c];
  }

  const int wave = tid >> 5, lane = tid & 31;
  const int m0 = wave * 16;
  const int mrow = lane & 15;
  const bool hi = lane >= 16;

  v8f acc[12];
  v8f zero = {};
  #pragma unroll
  for (int t = 0; t < 12; ++t) acc[t] = zero;

  for (int kk = 0; kk < H; kk += 32) {
    // stage V K-slice transposed: v4bf global reads, scattered b16 stores
    for (int i = tid; i < 32 * (C / 4); i += 256) {
      int r = i / 48, c4 = (i % 48) * 4;
      v4bf vv = *(const v4bf*)&Vp[vbase + (size_t)(kk + r) * vstr + c4];
      lVt[c4 + 0][r] = vv[0];
      lVt[c4 + 1][r] = vv[1];
      lVt[c4 + 2][r] = vv[2];
      lVt[c4 + 3][r] = vv[3];
    }
    if (kk + 32 < H)                              // overlap next slice fetch
      __builtin_prefetch(&Vp[vbase + (size_t)(kk + 32 + (tid >> 3)) * vstr], 0, 0);
    __syncthreads();

    v16bf a = frag_a(&lA[m0 + mrow][kk], lane);
    #pragma unroll
    for (int t = 0; t < 12; ++t) {
      v16bf bb = frag_b(&lVt[t * 16 + mrow][0], lane);
      acc[t] = wmma_bf16(a, bb, acc[t]);
    }
    __syncthreads();
  }

  #pragma unroll
  for (int t = 0; t < 12; ++t) {
    #pragma unroll
    for (int r = 0; r < 8; ++r) {
      int row = m0 + r + (hi ? 8 : 0);
      int col = t * 16 + mrow;
      size_t idx = vbase + (size_t)row * vstr + col;   // Acc shares V layout
      if (MODE == 0) Acc[idx] = acc[t][r];
      else           Acc[idx] += acc[t][r];
    }
  }
}

// ------------------------------------------------------------------------
// Kernel 5: out[b,c,h,w] = gamma * Acc[b, h*W+w, c] + x[b,c,h,w]
// ------------------------------------------------------------------------
__global__ __launch_bounds__(256)
void final_kernel(const float* __restrict__ x, const float* __restrict__ Acc,
                  const float* __restrict__ gamma, float* __restrict__ out) {
  const size_t idx = (size_t)blockIdx.x * 256 + threadIdx.x;
  const int w = (int)(idx & (W - 1));
  size_t t = idx >> 7;
  const int h = (int)(t & (H - 1));
  t >>= 7;
  const int c = (int)(t % C);
  const int b = (int)(t / C);
  const float g = gamma[0];
  const float a = Acc[((size_t)b * HW + (size_t)h * W + w) * C + c];
  out[idx] = fmaf(g, a, x[idx]);
}

// ------------------------------------------------------------------------
extern "C" void kernel_launch(void* const* d_in, const int* in_sizes, int n_in,
                              void* d_out, int out_size, void* d_ws, size_t ws_size,
                              hipStream_t stream) {
  const float* x  = (const float*)d_in[0];
  const float* Wq = (const float*)d_in[1];
  const float* bq = (const float*)d_in[2];
  const float* Wk = (const float*)d_in[3];
  const float* bk = (const float*)d_in[4];
  const float* Wv = (const float*)d_in[5];
  const float* bv = (const float*)d_in[6];
  const float* gm = (const float*)d_in[7];
  float* out = (float*)d_out;
  (void)in_sizes; (void)n_in; (void)out_size; (void)ws_size;

  char* ws = (char*)d_ws;
  size_t off = 0;
  auto carve = [&](size_t bytes) -> char* {
    char* p = ws + off;
    off += (bytes + 255) & ~(size_t)255;
    return p;
  };
  bf16*  Qp  = (bf16*) carve((size_t)B * HW * CQK  * sizeof(bf16));   // 16.8 MB
  bf16*  Kp  = (bf16*) carve((size_t)B * HW * CQK  * sizeof(bf16));   // 16.8 MB
  bf16*  Vp  = (bf16*) carve((size_t)B * HW * C    * sizeof(bf16));   // 50.3 MB
  bf16*  Att = (bf16*) carve((size_t)B * HW * NATT * sizeof(bf16));   // 67.1 MB
  float* Acc = (float*)carve((size_t)B * HW * C    * sizeof(float));  // 100.7 MB

  qkv_proj_kernel<<<dim3(HW / 64, 320 / 32, B), 256, 0, stream>>>(
      x, Wq, bq, Wk, bk, Wv, bv, Qp, Kp, Vp);
  gram_kernel<0><<<dim3(W, B), 256, 0, stream>>>(Qp, Kp, Att);
  gram_kernel<1><<<dim3(H, B), 256, 0, stream>>>(Qp, Kp, Att);
  softmax_kernel<<<dim3(B * HW / 8), 256, 0, stream>>>(Att);
  attn_out_kernel<0><<<dim3(W, B), 256, 0, stream>>>(Att, Vp, Acc);
  attn_out_kernel<1><<<dim3(H, B), 256, 0, stream>>>(Att, Vp, Acc);
  final_kernel<<<dim3((unsigned)((size_t)B * C * HW / 256)), 256, 0, stream>>>(
      x, Acc, gm, out);
}
